// GraphGatedGCNModel_15109694947694
// MI455X (gfx1250) — matrix operand order repaired and checked by reading, not compiled, not run on hardware
//
#include <hip/hip_runtime.h>

// ---------------------------------------------------------------------------
// GatedGCN (GNNome) forward for MI455X / gfx1250.
// GEMMs: v_wmma_f32_16x16x32_bf16 with weights staged into LDS via the
// Tensor Data Mover (tensor_load_to_lds + s_wait_tensorcnt) when available.
// Message passing: fused gather/BN-stat and sigmoid/scatter with f32 atomics.
// ---------------------------------------------------------------------------

#define HID 128
#define NNODES 50000
#define NEDGES 400000

typedef __attribute__((ext_vector_type(16))) __bf16 bf16x16;
typedef __attribute__((ext_vector_type(8)))  float  f32x8;
typedef __attribute__((ext_vector_type(4)))  unsigned int u32x4;
typedef __attribute__((ext_vector_type(8)))  int i32x8v;
typedef __attribute__((ext_vector_type(4)))  int i32x4v;

#if __has_builtin(__builtin_amdgcn_tensor_load_to_lds) && \
    __has_builtin(__builtin_amdgcn_s_wait_tensorcnt)
#define USE_TDM 1
#else
#define USE_TDM 0
#endif

#if USE_TDM
// ---------------------------------------------------------------------------
// TDM: DMA a contiguous block (n_dwords * 4 bytes) from global into LDS.
// D# built per CDNA5 ISA §8: 1-D tile of 4-byte elements,
// tensor_dim0 = tile_dim0 = n_dwords, tensor_dim1 = tile_dim1 = 1.
// Issue from one wave only (TDM ignores EXEC); caller barriers afterwards.
// This toolchain exposes the 6-arg builtin: (u32x4, i32x8, i32x4, i32x4,
// i32x8, i32 cpol) — trailing i32x8 group unused here (zero).
// ---------------------------------------------------------------------------
__device__ __forceinline__ void tdm_load_1d(const void* gsrc, void* lds_dst,
                                            unsigned n_dwords) {
  unsigned lds = (unsigned)(size_t)lds_dst;  // low 32 bits = LDS byte offset
  unsigned long long ga = (unsigned long long)(size_t)gsrc;
  u32x4 g0;
  g0[0] = 1u;                                            // count=1, user mode
  g0[1] = lds;                                           // lds_addr
  g0[2] = (unsigned)ga;                                  // global_addr[31:0]
  g0[3] = (unsigned)((ga >> 32) & 0x01FFFFFFu) | (2u << 30);  // ga[56:32], type=2
  i32x8v g1;
  g1[0] = (int)(2u << 16);                               // data_size=2 (4B)
  g1[1] = (int)((n_dwords & 0xFFFFu) << 16);             // tensor_dim0[15:0]
  g1[2] = (int)((n_dwords >> 16) & 0xFFFFu) | (1 << 16); // dim0[31:16], tensor_dim1=1
  g1[3] = (int)((n_dwords & 0xFFFFu) << 16);             // tile_dim0
  g1[4] = 1;                                             // tile_dim1=1, tile_dim2=0
  g1[5] = (int)n_dwords;                                 // tensor_dim0_stride[31:0]
  g1[6] = 0;                                             // stride hi / dim1_stride lo
  g1[7] = 0;
  i32x4v g2 = {1, 1, 0, 0};                              // tensor_dim2=1, dim3=1
  i32x4v g3 = {0, 0, 0, 0};
  i32x8v g4 = {0, 0, 0, 0, 0, 0, 0, 0};                  // unused trailing group
  __builtin_amdgcn_tensor_load_to_lds(g0, g1, g2, g3, g4, 0);
}
#endif

// Stage nbytes (multiple of 16) of packed weights into LDS; all threads call,
// barrier included.
__device__ __forceinline__ void stage_weights(const __bf16* __restrict__ Wp,
                                              __bf16* sw, unsigned nbytes) {
#if USE_TDM
  if ((threadIdx.x >> 5) == 0) {
    tdm_load_1d(Wp, sw, nbytes >> 2);
    __builtin_amdgcn_s_wait_tensorcnt(0);
  }
#else
  const uint4* gs = (const uint4*)Wp;
  uint4* ld = (uint4*)sw;
  unsigned n16 = nbytes >> 4;
  for (unsigned i = threadIdx.x; i < n16; i += blockDim.x) ld[i] = gs[i];
#endif
  __syncthreads();
}

// ---------------------------------------------------------------------------
// Weight packing: row-major f32 W[K][Nn] -> bf16 WMMA B-operand tiles.
// B layout (16-bit, K=32/tile): lane<16 -> K{0..7,16..23}, lane>=16 -> K{8..15,24..31},
// column N = lane%16.  Tile order: ktile*(Nn/16)+ntile, 32 lanes x 16 bf16.
// ---------------------------------------------------------------------------
__global__ __launch_bounds__(256)
void pack_wmma_b(const float* __restrict__ W, __bf16* __restrict__ out,
                 int K, int Nn) {
  int tiles_n = Nn >> 4;
  int nk      = K >> 5;
  int wid  = blockIdx.x * (blockDim.x >> 5) + (threadIdx.x >> 5);
  int lane = threadIdx.x & 31;
  if (wid >= nk * tiles_n) return;
  int t = wid / tiles_n;
  int j = wid - t * tiles_n;
  int n  = j * 16 + (lane & 15);
  int k0 = t * 32 + ((lane < 16) ? 0 : 8);
  __bf16* o = out + (((size_t)wid * 32 + lane) << 4);
#pragma unroll
  for (int i = 0; i < 16; ++i)
    o[i] = (__bf16)W[(size_t)(k0 + (i < 8 ? i : i + 8)) * Nn + n];
}

// ---------------------------------------------------------------------------
// Y[rows,128] = X[rows,128] @ W[128,128] + bias.  Weights (32KB packed bf16)
// staged in LDS by TDM; one wave per 16-row tile computes a 16x128 stripe.
// A fragment per ISA 16-bit A layout: lane<16 -> K{0..7,16..23}, lane>=16 ->
// K{8..15,24..31}, row M = lane%16.
// ---------------------------------------------------------------------------
__global__ __launch_bounds__(256)
void gemm128_bf16(const float* __restrict__ X, const __bf16* __restrict__ Wp,
                  const float* __restrict__ bias, float* __restrict__ Y,
                  int rows) {
  __shared__ __attribute__((aligned(32))) __bf16 sw[16384];  // 32 KB
  stage_weights(Wp, sw, 32768u);

  int wave = threadIdx.x >> 5;
  int lane = threadIdx.x & 31;
  int mt = blockIdx.x * 8 + wave;
  if (mt * 16 >= rows) return;
  int mrow  = mt * 16 + (lane & 15);
  int kbase = (lane < 16) ? 0 : 8;
  const bf16x16* swv = (const bf16x16*)sw;

  f32x8 c[8] = {};
#pragma unroll
  for (int t = 0; t < 4; ++t) {
    const float* ap = X + (size_t)mrow * HID + t * 32 + kbase;
    if (t < 3) __builtin_prefetch(ap + 32, 0, 0);
    bf16x16 a;
#pragma unroll
    for (int i = 0; i < 8; ++i) a[i] = (__bf16)ap[i];
#pragma unroll
    for (int i = 0; i < 8; ++i) a[8 + i] = (__bf16)ap[16 + i];
    bf16x16 breg[8];
#pragma unroll
    for (int j = 0; j < 8; ++j) breg[j] = swv[(t * 8 + j) * 32 + lane];
#pragma unroll
    for (int j = 0; j < 8; ++j)
      c[j] = __builtin_amdgcn_wmma_f32_16x16x32_bf16(
          false, a, false, breg[j], (short)0, c[j], false, false);
  }
  int mbase = mt * 16 + ((lane < 16) ? 0 : 8);
  int ncol  = lane & 15;
#pragma unroll
  for (int j = 0; j < 8; ++j) {
    int   n  = j * 16 + ncol;
    float bb = bias[n];
#pragma unroll
    for (int r = 0; r < 8; ++r)
      Y[(size_t)(mbase + r) * HID + n] = c[j][r] + bb;
  }
}

// ---------------------------------------------------------------------------
// Predictor: mid[E,64] = relu(cat(h[src],h[dst],e)[E,384] @ W1 + b1).
// W1 packed bf16 (48KB) staged in LDS; A gathered on the fly (each 32-wide
// K step lies entirely in one of the three sources since 128 % 32 == 0).
// ---------------------------------------------------------------------------
__global__ __launch_bounds__(256)
void pred_gemm_bf16(const float* __restrict__ h, const float* __restrict__ e,
                    const int* __restrict__ src, const int* __restrict__ dst,
                    const __bf16* __restrict__ Wp, const float* __restrict__ bias,
                    float* __restrict__ mid, int nedges) {
  __shared__ __attribute__((aligned(32))) __bf16 swp[24576];  // 48 KB
  stage_weights(Wp, swp, 49152u);

  int wave = threadIdx.x >> 5;
  int lane = threadIdx.x & 31;
  int mt = blockIdx.x * 8 + wave;
  if (mt * 16 >= nedges) return;
  int erow  = mt * 16 + (lane & 15);
  int kbase = (lane < 16) ? 0 : 8;
  int s = src[erow], d = dst[erow];
  const float* b0 = h + (size_t)s * HID;
  const float* b1 = h + (size_t)d * HID;
  const float* b2 = e + (size_t)erow * HID;
  const bf16x16* swv = (const bf16x16*)swp;

  f32x8 c[4] = {};
#pragma unroll
  for (int kt = 0; kt < 12; ++kt) {
    const float* bp = (kt < 4) ? b0 : (kt < 8) ? b1 : b2;
    const float* ap = bp + (kt & 3) * 32 + kbase;
    bf16x16 a;
#pragma unroll
    for (int i = 0; i < 8; ++i) a[i] = (__bf16)ap[i];
#pragma unroll
    for (int i = 0; i < 8; ++i) a[8 + i] = (__bf16)ap[16 + i];
    bf16x16 breg[4];
#pragma unroll
    for (int j = 0; j < 4; ++j) breg[j] = swv[(kt * 4 + j) * 32 + lane];
#pragma unroll
    for (int j = 0; j < 4; ++j)
      c[j] = __builtin_amdgcn_wmma_f32_16x16x32_bf16(
          false, a, false, breg[j], (short)0, c[j], false, false);
  }
  int mbase = mt * 16 + ((lane < 16) ? 0 : 8);
  int ncol  = lane & 15;
#pragma unroll
  for (int j = 0; j < 4; ++j) {
    int   n  = j * 16 + ncol;
    float bb = bias[n];
#pragma unroll
    for (int r = 0; r < 8; ++r) {
      float v = c[j][r] + bb;
      mid[(size_t)(mbase + r) * 64 + n] = v > 0.f ? v : 0.f;
    }
  }
}

// ---------------------------------------------------------------------------
// scores[E] = mid[E,64] @ W2[64] + b2
// ---------------------------------------------------------------------------
__global__ void score_out(const float* __restrict__ mid,
                          const float* __restrict__ W2,
                          const float* __restrict__ b2,
                          float* __restrict__ out, int E) {
  int ed = blockIdx.x * blockDim.x + threadIdx.x;
  if (ed >= E) return;
  const float* m = mid + (size_t)ed * 64;
  float acc = b2[0];
#pragma unroll
  for (int k = 0; k < 64; ++k) acc += m[k] * W2[k];
  out[ed] = acc;
}

// ---------------------------------------------------------------------------
// h = linear_pe(pe): K=18, one block per node, one thread per output feature.
// ---------------------------------------------------------------------------
__global__ __launch_bounds__(128)
void pe_embed(const float* __restrict__ pe, const float* __restrict__ W,
              const float* __restrict__ b, float* __restrict__ h, int N) {
  __shared__ float spe[18];
  int node = blockIdx.x;
  if (node >= N) return;
  if (threadIdx.x < 18) spe[threadIdx.x] = pe[(size_t)node * 18 + threadIdx.x];
  __syncthreads();
  int f = threadIdx.x;
  float acc = b[f];
#pragma unroll
  for (int k = 0; k < 18; ++k) acc += spe[k] * W[k * HID + f];
  h[(size_t)node * HID + f] = acc;
}

// ---------------------------------------------------------------------------
// e = lin2(relu(lin1(e2))): 2 -> 16 -> 128, one block per edge.
// ---------------------------------------------------------------------------
__global__ __launch_bounds__(128)
void edge_embed(const float* __restrict__ e2, const float* __restrict__ W1,
                const float* __restrict__ b1, const float* __restrict__ W2,
                const float* __restrict__ b2, float* __restrict__ e, int E) {
  __shared__ float hid[16];
  int ed = blockIdx.x;
  if (ed >= E) return;
  if (threadIdx.x < 16) {
    float a0 = e2[(size_t)ed * 2], a1 = e2[(size_t)ed * 2 + 1];
    float v = b1[threadIdx.x] + a0 * W1[threadIdx.x] + a1 * W1[16 + threadIdx.x];
    hid[threadIdx.x] = v > 0.f ? v : 0.f;
  }
  __syncthreads();
  float acc = b2[threadIdx.x];
#pragma unroll
  for (int k = 0; k < 16; ++k) acc += hid[k] * W2[k * HID + threadIdx.x];
  e[(size_t)ed * HID + threadIdx.x] = acc;
}

__global__ void zero_f32(float* __restrict__ p, long long n) {
  long long i = (long long)blockIdx.x * blockDim.x + threadIdx.x;
  if (i < n) p[i] = 0.f;
}

// ---------------------------------------------------------------------------
// ehat += B1h[src] + B2h[dst]; per-feature BN sums with one atomicAdd per
// (block, feature). blockDim = 128 (feature = tid).
// ---------------------------------------------------------------------------
__global__ __launch_bounds__(128)
void edge_ehat_stats(float* __restrict__ ehat, const float* __restrict__ B1h,
                     const float* __restrict__ B2h, const int* __restrict__ src,
                     const int* __restrict__ dst, float* __restrict__ sums,
                     float* __restrict__ sumsq, int E, int chunk) {
  int f  = threadIdx.x;
  int e0 = blockIdx.x * chunk;
  int e1 = e0 + chunk; if (e1 > E) e1 = E;
  float s = 0.f, s2 = 0.f;
  for (int ed = e0; ed < e1; ++ed) {
    int sv = src[ed], dv = dst[ed];
    size_t i = (size_t)ed * HID + f;
    float v = ehat[i] + B1h[(size_t)sv * HID + f] + B2h[(size_t)dv * HID + f];
    ehat[i] = v;
    s += v; s2 += v * v;
  }
  atomicAdd(&sums[f], s);
  atomicAdd(&sumsq[f], s2);
}

__global__ void bn_finalize(const float* __restrict__ sums,
                            const float* __restrict__ sumsq,
                            const float* __restrict__ g,
                            const float* __restrict__ b,
                            float* __restrict__ scale,
                            float* __restrict__ shift, float count) {
  int f = threadIdx.x;
  float mu  = sums[f] / count;
  float var = sumsq[f] / count - mu * mu;
  float sc  = g[f] * rsqrtf(var + 1e-5f);
  scale[f] = sc;
  shift[f] = b[f] - mu * sc;
}

// ---------------------------------------------------------------------------
// e += relu(BN(ehat)); sigma = sigmoid(e); scatter 4 segment sums (atomics).
// ---------------------------------------------------------------------------
__global__ void edge_update_scatter(float* __restrict__ e,
                                    const float* __restrict__ ehat,
                                    const float* __restrict__ scale,
                                    const float* __restrict__ shift,
                                    const float* __restrict__ A2h,
                                    const float* __restrict__ A3h,
                                    const int* __restrict__ src,
                                    const int* __restrict__ dst,
                                    float* __restrict__ nf, float* __restrict__ df,
                                    float* __restrict__ nb, float* __restrict__ db,
                                    int E) {
  size_t idx = (size_t)blockIdx.x * blockDim.x + threadIdx.x;
  if (idx >= (size_t)E * HID) return;
  int ed = (int)(idx >> 7);
  int f  = (int)(idx & 127);
  int sv = src[ed], dv = dst[ed];
  float v  = ehat[idx] * scale[f] + shift[f];
  float en = e[idx] + (v > 0.f ? v : 0.f);
  e[idx] = en;
  float sg = 1.f / (1.f + __expf(-en));
  atomicAdd(&nf[(size_t)dv * HID + f], sg * A2h[(size_t)sv * HID + f]);
  atomicAdd(&df[(size_t)dv * HID + f], sg);
  atomicAdd(&nb[(size_t)sv * HID + f], sg * A3h[(size_t)dv * HID + f]);
  atomicAdd(&db[(size_t)sv * HID + f], sg);
}

__global__ __launch_bounds__(128)
void node_hhat_stats(float* __restrict__ hhat, const float* __restrict__ A1h,
                     const float* __restrict__ nf, const float* __restrict__ df,
                     const float* __restrict__ nb, const float* __restrict__ db,
                     float* __restrict__ sums, float* __restrict__ sumsq,
                     int N, int chunk) {
  int f  = threadIdx.x;
  int n0 = blockIdx.x * chunk;
  int n1 = n0 + chunk; if (n1 > N) n1 = N;
  float s = 0.f, s2 = 0.f;
  for (int n = n0; n < n1; ++n) {
    size_t i = (size_t)n * HID + f;
    float v = A1h[i] + nf[i] / (df[i] + 1e-6f) + nb[i] / (db[i] + 1e-6f);
    hhat[i] = v;
    s += v; s2 += v * v;
  }
  atomicAdd(&sums[f], s);
  atomicAdd(&sumsq[f], s2);
}

__global__ void node_update(float* __restrict__ h, const float* __restrict__ hhat,
                            const float* __restrict__ scale,
                            const float* __restrict__ shift, int N) {
  size_t idx = (size_t)blockIdx.x * blockDim.x + threadIdx.x;
  if (idx >= (size_t)N * HID) return;
  int f = (int)(idx & 127);
  float v = hhat[idx] * scale[f] + shift[f];
  h[idx] += (v > 0.f ? v : 0.f);
}

// ---------------------------------------------------------------------------
// Launch. Input flat order = setup_inputs() insertion order, recursively.
// ---------------------------------------------------------------------------
extern "C" void kernel_launch(void* const* d_in, const int* in_sizes, int n_in,
                              void* d_out, int out_size, void* d_ws, size_t ws_size,
                              hipStream_t stream) {
  const int N = NNODES, E = NEDGES;
  auto F = [&](int i) { return (const float*)d_in[i]; };
  // params: 0..9 globals, 10 + 16*L per layer, then x(74), e(75), pe(76), src(77), dst(78)
  const float* e_in  = F(75);
  const float* pe_in = F(76);
  const int* src = (const int*)d_in[77];
  const int* dst = (const int*)d_in[78];

  char* w = (char*)d_ws;
  auto carve = [&](size_t bytes) {
    void* p = (void*)w;
    w += (bytes + 255) & ~(size_t)255;
    return p;
  };
  float* h    = (float*)carve((size_t)N * HID * 4);
  float* t1   = (float*)carve((size_t)N * HID * 4);   // A1h
  float* t2   = (float*)carve((size_t)N * HID * 4);   // A2h
  float* t3   = (float*)carve((size_t)N * HID * 4);   // A3h
  float* t4   = (float*)carve((size_t)N * HID * 4);   // B1h / hhat
  float* t5   = (float*)carve((size_t)N * HID * 4);   // B2h
  float* e    = (float*)carve((size_t)E * HID * 4);
  float* ehat = (float*)carve((size_t)E * HID * 4);   // also mid[E,64] for predictor
  float* nf   = (float*)carve((size_t)4 * N * HID * 4);
  float* df = nf + (size_t)N * HID;
  float* nb = df + (size_t)N * HID;
  float* db = nb + (size_t)N * HID;
  float* sums  = (float*)carve(4 * HID * 4);
  float* sumsq = sums + HID;
  float* scale = sums + 2 * HID;
  float* shift = sums + 3 * HID;
  __bf16* pk = (__bf16*)carve(((size_t)24 * 128 * 128 + 384 * 64) * 2);
  __bf16* pkPred = pk + (size_t)24 * 128 * 128;

  auto pack = [&](const float* W, __bf16* out, int K, int Nn) {
    int waves = (K / 32) * (Nn / 16);
    int blocks = (waves * 32 + 255) / 256;
    pack_wmma_b<<<blocks, 256, 0, stream>>>(W, out, K, Nn);
  };
  for (int L = 0; L < 4; ++L) {
    int b = 10 + 16 * L;
    for (int m = 0; m < 6; ++m)   // A1,A2,A3,B1,B2,B3 weights at b+0,2,4,6,8,10
      pack(F(b + 2 * m), pk + ((size_t)L * 6 + m) * 128 * 128, 128, 128);
  }
  pack(F(6), pkPred, 384, 64);    // pred_W1

  pe_embed<<<N, 128, 0, stream>>>(pe_in, F(0), F(1), h, N);
  edge_embed<<<E, 128, 0, stream>>>(e_in, F(2), F(3), F(4), F(5), e, E);

  const int mtN = N / 16, mtE = E / 16;  // 3125, 25000 (both exact)
  for (int L = 0; L < 4; ++L) {
    int b = 10 + 16 * L;
    __bf16* pkL = pk + (size_t)L * 6 * 128 * 128;
    gemm128_bf16<<<(mtN + 7) / 8, 256, 0, stream>>>(h, pkL + 0 * 128 * 128, F(b + 1), t1, N);
    gemm128_bf16<<<(mtN + 7) / 8, 256, 0, stream>>>(h, pkL + 1 * 128 * 128, F(b + 3), t2, N);
    gemm128_bf16<<<(mtN + 7) / 8, 256, 0, stream>>>(h, pkL + 2 * 128 * 128, F(b + 5), t3, N);
    gemm128_bf16<<<(mtN + 7) / 8, 256, 0, stream>>>(h, pkL + 3 * 128 * 128, F(b + 7), t4, N);
    gemm128_bf16<<<(mtN + 7) / 8, 256, 0, stream>>>(h, pkL + 4 * 128 * 128, F(b + 9), t5, N);
    gemm128_bf16<<<(mtE + 7) / 8, 256, 0, stream>>>(e, pkL + 5 * 128 * 128, F(b + 11), ehat, E);

    zero_f32<<<1, 256, 0, stream>>>(sums, 256);
    edge_ehat_stats<<<(E + 255) / 256, 128, 0, stream>>>(ehat, t4, t5, src, dst,
                                                         sums, sumsq, E, 256);
    bn_finalize<<<1, 128, 0, stream>>>(sums, sumsq, F(b + 14), F(b + 15),
                                       scale, shift, (float)E);

    zero_f32<<<(int)(((size_t)4 * N * HID + 255) / 256), 256, 0, stream>>>(
        nf, (long long)4 * N * HID);
    edge_update_scatter<<<(int)(((size_t)E * HID + 255) / 256), 256, 0, stream>>>(
        e, ehat, scale, shift, t2, t3, src, dst, nf, df, nb, db, E);

    zero_f32<<<1, 256, 0, stream>>>(sums, 256);
    node_hhat_stats<<<(N + 255) / 256, 128, 0, stream>>>(t4, t1, nf, df, nb, db,
                                                         sums, sumsq, N, 256);
    bn_finalize<<<1, 128, 0, stream>>>(sums, sumsq, F(b + 12), F(b + 13),
                                       scale, shift, (float)N);
    node_update<<<(int)(((size_t)N * HID + 255) / 256), 256, 0, stream>>>(h, t4, scale,
                                                                          shift, N);
  }

  float* mid = ehat;  // reuse [E,128] region for [E,64] predictor hidden
  pred_gemm_bf16<<<(mtE + 7) / 8, 256, 0, stream>>>(h, e, src, dst, pkPred, F(7), mid, E);
  score_out<<<(E + 255) / 256, 256, 0, stream>>>(mid, F(8), F(9), (float*)d_out, E);
}
